// TimeRatingAwareContentModel_35218731827259
// MI455X (gfx1250) — matrix operand order repaired and checked by reading, not compiled, not run on hardware
//
#include <hip/hip_runtime.h>
#include <hip/hip_bf16.h>
#include <math.h>

// ---------------------------------------------------------------------------
// Model dims
#define BB   1024
#define SS   200
#define VV   100000
#define EE   64
#define GG   20
#define HH   128
#define DP   128          // D=100 padded to 128 (zeros)
#define G4   512          // 4*H
// ---------------------------------------------------------------------------

typedef __bf16 bf16;
typedef __attribute__((ext_vector_type(16))) __bf16 v16bf;
typedef __attribute__((ext_vector_type(8)))  float  v8f;

// ---- WMMA fragment loaders (CDNA5 16x16x32 bf16 layouts, wave32) ----------
// A (MxK = 16x32): lane holds row (lane&15); half0 -> K {k0..k0+7, k0+16..k0+23},
//                  half1 -> K {k0+8..k0+15, k0+24..k0+31}
__device__ __forceinline__ v16bf load_a_frag(const bf16* __restrict__ base, int ld,
                                             int row0, int k0, int lane) {
    int half = lane >> 4;
    int r    = lane & 15;
    const bf16* p = base + (size_t)(row0 + r) * ld + k0 + half * 8;
    v16bf v;
    ((uint4*)&v)[0] = *(const uint4*)(p);        // K k0+half*8   .. +7
    ((uint4*)&v)[1] = *(const uint4*)(p + 16);   // K k0+16+half*8.. +7
    return v;
}
// B (KxN = 32x16), weights stored transposed Wt[n][k]: lane holds column (lane&15);
// half0 -> K {k0..k0+7, k0+16..k0+23}, half1 -> K {k0+8..k0+15, k0+24..k0+31}
__device__ __forceinline__ v16bf load_b_frag(const bf16* __restrict__ base, int ld,
                                             int n0, int k0, int lane) {
    int half = lane >> 4;
    int n    = lane & 15;
    const bf16* p = base + (size_t)(n0 + n) * ld + k0 + half * 8;
    v16bf v;
    ((uint4*)&v)[0] = *(const uint4*)(p);
    ((uint4*)&v)[1] = *(const uint4*)(p + 16);
    return v;
}
__device__ __forceinline__ v8f wmma_bf16(v16bf a, v16bf b, v8f c) {
    return __builtin_amdgcn_wmma_f32_16x16x32_bf16(false, a, false, b,
                                                   (short)0, c, false, false);
}
__device__ __forceinline__ float sigm(float x) { return 1.0f / (1.0f + __expf(-x)); }
// saturating fast tanh via hardware v_exp_f32 (correct limits at +/-inf)
__device__ __forceinline__ float tanh_fast(float x) {
    return 2.0f / (1.0f + __expf(-2.0f * x)) - 1.0f;
}

// ---------------------------------------------------------------------------
// Prep: bf16-pack W_ih (pad K 100->128), W_hh; combine biases; transpose W1.
__global__ void k_prep_small(const float* __restrict__ W_ih, const float* __restrict__ W_hh,
                             const float* __restrict__ b_ih, const float* __restrict__ b_hh,
                             const float* __restrict__ W1,
                             bf16* __restrict__ Wih_bf, bf16* __restrict__ Whh_bf,
                             float* __restrict__ bias, bf16* __restrict__ W1t) {
    int idx = blockIdx.x * blockDim.x + threadIdx.x;   // 65536 threads
    if (idx < G4 * DP) {
        int n = idx >> 7, k = idx & 127;
        Wih_bf[idx] = (bf16)((k < 100) ? W_ih[n * 100 + k] : 0.0f);
        Whh_bf[idx] = (bf16)W_hh[idx];                 // W_hh is (512,128) row-major
    }
    if (idx < G4) bias[idx] = b_ih[idx] + b_hh[idx];
    if (idx < 256 * HH) {                              // W1t[n][k] = W1[k][n], W1 (128,256)
        int n = idx >> 7, k = idx & 127;
        W1t[idx] = (bf16)W1[k * 256 + n];
    }
}

// Tiled transpose: W2 (256, V) f32 -> W2t (V, 256) bf16
__global__ void k_transpose_w2(const float* __restrict__ W2, bf16* __restrict__ W2t) {
    __shared__ float tile[32][33];
    int n0 = blockIdx.x * 32, k0 = blockIdx.y * 32;
    int tx = threadIdx.x, ty = threadIdx.y;
    tile[ty][tx] = W2[(size_t)(k0 + ty) * VV + n0 + tx];
    __syncthreads();
    W2t[(size_t)(n0 + ty) * 256 + k0 + tx] = (bf16)tile[tx][ty];
}

// ---------------------------------------------------------------------------
// Build features in time-major row order r' = s*B + b:
// X_bf[r'][128] = [emb(64) | genre(20) | r(8) | t(8) | 0(28)]
__global__ void k_pack_x(const int* __restrict__ movie, const float* __restrict__ rating,
                         const float* __restrict__ time_, const float* __restrict__ emb,
                         const float* __restrict__ genre, const float* __restrict__ Wr,
                         const float* __restrict__ br, const float* __restrict__ Wt,
                         const float* __restrict__ bt, bf16* __restrict__ X) {
    int rp = blockIdx.x;           // r' = s*B + b
    int t  = threadIdx.x;          // 0 .. 127
    int s  = rp >> 10;             // / B
    int b  = rp & (BB - 1);
    int ridx = b * SS + s;         // index into [B,S] inputs
    int mv = movie[ridx];
    float v = 0.0f;
    if (t < 64) {
        v = emb[(size_t)mv * EE + t];
    } else if (t < 84) {
        int g = t - 64;
        int mc = mv < VV ? mv : (VV - 1);
        v = (mv < VV) ? genre[(size_t)mc * GG + g] : 0.0f;
    } else if (t < 92) {
        int j = t - 84;
        v = rating[ridx] * Wr[j] + br[j];
    } else if (t < 100) {
        int j = t - 92;
        v = time_[ridx] * Wt[j] + bt[j];
    }
    X[(size_t)rp * DP + t] = (bf16)v;
}

// ---------------------------------------------------------------------------
// Input GEMM: gates_in = X @ Wih^T + (b_ih+b_hh).  M=B*S (time-major), N=512, K=128.
// Output stored in WMMA *fragment layout*: G[(mt*32 + nt)*256 + lane*8 + i]
// so the LSTM kernel re-loads each C tile as two b128s per lane.
__global__ void k_input_gemm(const bf16* __restrict__ X, const bf16* __restrict__ Wih,
                             const float* __restrict__ bias, float* __restrict__ G) {
    int lane = threadIdx.x & 31, w = threadIdx.x >> 5;
    int mt = blockIdx.x;
    int m0 = mt * 16;
    int col = lane & 15;
    v8f acc[4];
#pragma unroll
    for (int j = 0; j < 4; j++) {
        float bv = bias[w * 64 + j * 16 + col];
#pragma unroll
        for (int i = 0; i < 8; i++) acc[j][i] = bv;
    }
#pragma unroll
    for (int kc = 0; kc < 4; kc++) {
        v16bf a = load_a_frag(X, DP, m0, kc * 32, lane);
#pragma unroll
        for (int j = 0; j < 4; j++) {
            v16bf b = load_b_frag(Wih, DP, w * 64 + j * 16, kc * 32, lane);
            acc[j] = wmma_bf16(a, b, acc[j]);
        }
    }
#pragma unroll
    for (int j = 0; j < 4; j++) {
        int nt = w * 4 + j;
        float* p = G + ((size_t)mt * 32 + nt) * 256 + lane * 8;
        *(float4*)(p)     = ((float4*)&acc[j])[0];
        *(float4*)(p + 4) = ((float4*)&acc[j])[1];
    }
}

// ---------------------------------------------------------------------------
// LSTM recurrence. 64 blocks x 16 batch rows, 8 waves. W_hh slice lives in VGPRs.
// Gate-input fragments for step s+1 are prefetched during step s's compute.
__global__ void __launch_bounds__(256) k_lstm(const bf16* __restrict__ Whh,
                                              const float* __restrict__ G,
                                              bf16* __restrict__ h_out) {
    __shared__ bf16  sH[16 * 136];   // h tile, bf16, padded stride (272B, 16B aligned)
    __shared__ float sC[16 * 128];   // c tile
    __shared__ float sG[16 * 520];   // gate pre-activations (padded stride)

    int tid = threadIdx.x, lane = tid & 31, w = tid >> 5;
    int col = lane & 15, half = lane >> 4;
    int bt0 = blockIdx.x * 16;

    for (int i = tid; i < 16 * 136; i += 256) sH[i] = (bf16)0.0f;
    for (int i = tid; i < 16 * 128; i += 256) sC[i] = 0.0f;

    // Resident W_hh B-fragments: wave w owns N in [w*64, w*64+64)  (128 VGPRs/lane)
    v16bf bfrag[4][4];
#pragma unroll
    for (int j = 0; j < 4; j++)
#pragma unroll
        for (int kc = 0; kc < 4; kc++)
            bfrag[j][kc] = load_b_frag(Whh, HH, w * 64 + j * 16, kc * 32, lane);
    __syncthreads();

    int m_ew  = tid >> 4;          // elementwise row (0..15)
    int c0_ew = (tid & 15) * 8;    // elementwise col base (8 cols/thread)

    // fragment base for this (block, wave, lane): tile mt(s) = s*64 + blockIdx.x
    auto gptr = [&](int s, int j) -> const float* {
        size_t mt = (size_t)s * 64 + blockIdx.x;
        return G + (mt * 32 + (w * 4 + j)) * 256 + lane * 8;
    };

    // prefetch step 0
    v8f gnext[4];
#pragma unroll
    for (int j = 0; j < 4; j++) {
        const float* p = gptr(0, j);
        ((float4*)&gnext[j])[0] = *(const float4*)(p);
        ((float4*)&gnext[j])[1] = *(const float4*)(p + 4);
    }

    for (int s = 0; s < SS; s++) {
        v8f acc[4];
#pragma unroll
        for (int j = 0; j < 4; j++) acc[j] = gnext[j];
        if (s + 1 < SS) {          // uniform branch: prefetch next step's C tiles
#pragma unroll
            for (int j = 0; j < 4; j++) {
                const float* p = gptr(s + 1, j);
                ((float4*)&gnext[j])[0] = *(const float4*)(p);
                ((float4*)&gnext[j])[1] = *(const float4*)(p + 4);
            }
        }
        // gates += h @ W_hh^T
#pragma unroll
        for (int kc = 0; kc < 4; kc++) {
            v16bf a = load_a_frag(sH, 136, 0, kc * 32, lane);
#pragma unroll
            for (int j = 0; j < 4; j++) acc[j] = wmma_bf16(a, bfrag[j][kc], acc[j]);
        }
#pragma unroll
        for (int j = 0; j < 4; j++)
#pragma unroll
            for (int i = 0; i < 8; i++)
                sG[(i + 8 * half) * 520 + w * 64 + j * 16 + col] = acc[j][i];
        __syncthreads();

        // elementwise gate math: i,f,g,o chunks of 128 (torch order)
#pragma unroll
        for (int jj = 0; jj < 8; jj++) {
            int cc = c0_ew + jj;
            float ig = sigm(sG[m_ew * 520 + cc]);
            float fg = sigm(sG[m_ew * 520 + 128 + cc]);
            float gg = tanh_fast(sG[m_ew * 520 + 256 + cc]);
            float og = sigm(sG[m_ew * 520 + 384 + cc]);
            float c  = fg * sC[m_ew * 128 + cc] + ig * gg;
            sC[m_ew * 128 + cc] = c;
            sH[m_ew * 136 + cc] = (bf16)(og * tanh_fast(c));
        }
        __syncthreads();
    }
    for (int i = tid; i < 16 * 128; i += 256) {
        int m = i >> 7, jj = i & 127;
        h_out[(size_t)(bt0 + m) * HH + jj] = sH[m * 136 + jj];
    }
}

// ---------------------------------------------------------------------------
// d1 = relu(h @ W1 + b1) -> bf16 [1024][256].  4 waves/block, 64 blocks.
__global__ void k_mlp1(const bf16* __restrict__ h, const bf16* __restrict__ W1t,
                       const float* __restrict__ b1, bf16* __restrict__ d1) {
    int lane = threadIdx.x & 31, w = threadIdx.x >> 5;
    int m0 = blockIdx.x * 16;
    int col = lane & 15, half = lane >> 4;
    v8f acc[4];
#pragma unroll
    for (int j = 0; j < 4; j++) {
        float bv = b1[w * 64 + j * 16 + col];
#pragma unroll
        for (int i = 0; i < 8; i++) acc[j][i] = bv;
    }
#pragma unroll
    for (int kc = 0; kc < 4; kc++) {
        v16bf a = load_a_frag(h, HH, m0, kc * 32, lane);
#pragma unroll
        for (int j = 0; j < 4; j++) {
            v16bf b = load_b_frag(W1t, HH, w * 64 + j * 16, kc * 32, lane);
            acc[j] = wmma_bf16(a, b, acc[j]);
        }
    }
#pragma unroll
    for (int j = 0; j < 4; j++)
#pragma unroll
        for (int i = 0; i < 8; i++) {
            float v = acc[j][i];
            v = v > 0.0f ? v : 0.0f;
            d1[(size_t)(m0 + i + 8 * half) * 256 + w * 64 + j * 16 + col] = (bf16)v;
        }
}

// ---------------------------------------------------------------------------
// logits = d1 @ W2 + b2.  M=1024, N=100000 (6250 tiles of 16), K=256.
// grid (64, 196); block 8 waves; wave w handles tiles blockIdx.y*32 + w*4 + j.
__global__ void __launch_bounds__(256) k_logits(const bf16* __restrict__ d1,
                                                const bf16* __restrict__ W2t,
                                                const float* __restrict__ b2,
                                                float* __restrict__ out) {
    int lane = threadIdx.x & 31, w = threadIdx.x >> 5;
    int m0 = blockIdx.x * 16;
    int col = lane & 15, half = lane >> 4;
    v16bf a[8];
#pragma unroll
    for (int kc = 0; kc < 8; kc++) a[kc] = load_a_frag(d1, 256, m0, kc * 32, lane);
#pragma unroll
    for (int j = 0; j < 4; j++) {
        int nt = blockIdx.y * 32 + w * 4 + j;     // wave-uniform -> EXEC stays full
        if (nt >= VV / 16) continue;
        int n0 = nt * 16;
        v8f acc;
        float bv = b2[n0 + col];
#pragma unroll
        for (int i = 0; i < 8; i++) acc[i] = bv;
#pragma unroll
        for (int kc = 0; kc < 8; kc++) {
            v16bf b = load_b_frag(W2t, 256, n0, kc * 32, lane);
            acc = wmma_bf16(a[kc], b, acc);
        }
#pragma unroll
        for (int i = 0; i < 8; i++)
            out[(size_t)(m0 + i + 8 * half) * VV + n0 + col] = acc[i];
    }
}

// ---------------------------------------------------------------------------
extern "C" void kernel_launch(void* const* d_in, const int* in_sizes, int n_in,
                              void* d_out, int out_size, void* d_ws, size_t ws_size,
                              hipStream_t stream) {
    const int*   movie  = (const int*)  d_in[0];
    const float* rating = (const float*)d_in[1];
    const float* time_  = (const float*)d_in[2];
    const float* emb    = (const float*)d_in[3];
    const float* genre  = (const float*)d_in[4];
    const float* W_r    = (const float*)d_in[5];
    const float* b_r    = (const float*)d_in[6];
    const float* W_t    = (const float*)d_in[7];
    const float* b_t    = (const float*)d_in[8];
    const float* W_ih   = (const float*)d_in[9];
    const float* W_hh   = (const float*)d_in[10];
    const float* b_ih   = (const float*)d_in[11];
    const float* b_hh   = (const float*)d_in[12];
    const float* W1     = (const float*)d_in[13];
    const float* b1     = (const float*)d_in[14];
    const float* W2     = (const float*)d_in[15];
    const float* b2     = (const float*)d_in[16];
    float* out = (float*)d_out;

    // workspace carve-up (256B aligned)
    char* ws = (char*)d_ws;
    size_t off = 0;
    auto carve = [&](size_t bytes) { size_t o = off; off = (off + bytes + 255) & ~(size_t)255; return o; };
    bf16*  Wih_bf = (bf16*) (ws + carve((size_t)G4 * DP * 2));        // 128 KB
    bf16*  Whh_bf = (bf16*) (ws + carve((size_t)G4 * DP * 2));        // 128 KB
    float* biasC  = (float*)(ws + carve((size_t)G4 * 4));             // 2 KB
    bf16*  W1t    = (bf16*) (ws + carve((size_t)256 * HH * 2));       // 64 KB
    bf16*  W2t    = (bf16*) (ws + carve((size_t)VV * 256 * 2));       // 51.2 MB
    bf16*  Xbf    = (bf16*) (ws + carve((size_t)BB * SS * DP * 2));   // 52.4 MB
    float* Gws    = (float*)(ws + carve((size_t)BB * SS * G4 * 4));   // 419 MB (fragment layout)
    bf16*  h_bf   = (bf16*) (ws + carve((size_t)BB * HH * 2));        // 256 KB
    bf16*  d1bf   = (bf16*) (ws + carve((size_t)BB * 256 * 2));       // 512 KB

    k_prep_small<<<256, 256, 0, stream>>>(W_ih, W_hh, b_ih, b_hh, W1,
                                          Wih_bf, Whh_bf, biasC, W1t);
    k_transpose_w2<<<dim3(VV / 32, 256 / 32), dim3(32, 32), 0, stream>>>(W2, W2t);
    k_pack_x<<<BB * SS, 128, 0, stream>>>(movie, rating, time_, emb, genre,
                                          W_r, b_r, W_t, b_t, Xbf);
    k_input_gemm<<<(BB * SS) / 16, 256, 0, stream>>>(Xbf, Wih_bf, biasC, Gws);
    k_lstm<<<BB / 16, 256, 0, stream>>>(Whh_bf, Gws, h_bf);
    k_mlp1<<<BB / 16, 128, 0, stream>>>(h_bf, W1t, b1, d1bf);
    k_logits<<<dim3(BB / 16, 196), 256, 0, stream>>>(d1bf, W2t, b2, out);
}
